// SequentialRNNGNN_76510547411598
// MI455X (gfx1250) — matrix operand order repaired and checked by reading, not compiled, not run on hardware
//
#include <hip/hip_runtime.h>

#define NN   8192
#define EE   32768
#define IN_  64
#define H1_  128
#define RH_  128
#define H2_  128
#define GG   32

typedef _Float16 h16;
typedef __attribute__((ext_vector_type(8)))  _Float16 v8h;
typedef __attribute__((ext_vector_type(16))) _Float16 v16h;
typedef __attribute__((ext_vector_type(8)))  float    v8f;
typedef __attribute__((ext_vector_type(4)))  int      v4i;

// CDNA5 async global->LDS copy path (guarded; falls back to prefetch)
#if defined(__gfx1250__) && __has_builtin(__builtin_amdgcn_global_load_async_to_lds_b128)
#define ASYNC_LDS 1
#else
#define ASYNC_LDS 0
#endif

__device__ __forceinline__ void async_copy16(const float* g, float* l) {
#if ASYNC_LDS
  // builtin signature: (v4i AS1*, v4i AS3*, imm int offset, imm int cpol)
  const v4i* gvc = (const v4i*)g;                 // bitcast in generic AS
  v4i*       gvv = const_cast<v4i*>(gvc);
  v4i*       lvv = (v4i*)l;
  __builtin_amdgcn_global_load_async_to_lds_b128(
      (__attribute__((address_space(1))) v4i*)gvv,
      (__attribute__((address_space(3))) v4i*)lvv, 0, 0);
#else
  (void)g; (void)l;
#endif
}
__device__ __forceinline__ void wait_async_zero() {
#if ASYNC_LDS
#if __has_builtin(__builtin_amdgcn_s_wait_asynccnt)
  __builtin_amdgcn_s_wait_asynccnt(0);
#else
  asm volatile("s_wait_asynccnt 0x0" ::: "memory");
#endif
#endif
}

__device__ __forceinline__ float leaky02(float v) { return v > 0.f ? v : 0.2f * v; }

__device__ __forceinline__ float atomicMaxF(float* addr, float val) {
  int* ai = (int*)addr;
  int old = __float_as_int(*addr);
  while (__int_as_float(old) < val) {
    int assumed = old;
    old = atomicCAS(ai, assumed, __float_as_int(val));
    if (old == assumed) break;
  }
  return __int_as_float(old);
}

// ---------------------------------------------------------------------------
// WMMA GEMM: C[M,N] = A[M,K](f16) * Bt[N,K](f16)^T, f32 accumulate.
// One wave per 16x64 output STRIP (4 accumulators): A tile loaded once per
// K-step and reused across 4 WMMAs -> 2.5 b128 loads per WMMA instead of 4.
// K % 32 == 0, M % 16 == 0, N % 64 == 0.
// mode 0: C = acc + p0[n]
// mode 1: C = relu((acc + p0[n] - p1[n])*rsqrt(p2[n]+eps)*p3[n] + p4[n])
// mode 2: C = acc
// ---------------------------------------------------------------------------
__global__ void wmma_gemm_kernel(const _Float16* __restrict__ A,
                                 const _Float16* __restrict__ Bt,
                                 float* __restrict__ C,
                                 int M, int N, int K, int mode,
                                 const float* __restrict__ p0,
                                 const float* __restrict__ p1,
                                 const float* __restrict__ p2,
                                 const float* __restrict__ p3,
                                 const float* __restrict__ p4)
{
  const int lane    = threadIdx.x & 31;
  const int wave    = threadIdx.x >> 5;
  const int stripsN = N >> 6;                 // strips of 4 N-tiles
  const int tilesM  = M >> 4;
  const int strip   = blockIdx.x * (blockDim.x >> 5) + wave;
  if (strip >= tilesM * stripsN) return;
  const int tm   = strip / stripsN;
  const int sn   = strip % stripsN;
  const int half = lane >> 4;                 // 0: lanes 0-15, 1: lanes 16-31
  const int l15  = lane & 15;
  const int arow = tm * 16 + l15;

  const _Float16* arowp = A  + (size_t)arow * K + half * 16;
  const _Float16* bbase = Bt + (size_t)(sn * 64 + l15) * K + half * 16;

  v8f acc[4] = {};
  #pragma unroll 2
  for (int kb = 0; kb < K; kb += 32) {
    // A tile 16x32: halves i=0..15 -> K=kb+16*half+i
    v8h alo = *(const v8h*)(arowp + kb);
    v8h ahi = *(const v8h*)(arowp + kb + 8);
    v16h av = __builtin_shufflevector(alo, ahi,
                0, 1, 2, 3, 4, 5, 6, 7, 8, 9, 10, 11, 12, 13, 14, 15);
    // 4 B tiles 32x16: halves i -> K=kb+16*half+i, columns sn*64 + j*16 + l15
    v16h b0 = *(const v16h*)(bbase + kb);
    v16h b1 = *(const v16h*)(bbase + (size_t)16 * K + kb);
    v16h b2 = *(const v16h*)(bbase + (size_t)32 * K + kb);
    v16h b3 = *(const v16h*)(bbase + (size_t)48 * K + kb);
    acc[0] = __builtin_amdgcn_wmma_f32_16x16x32_f16(false, av, false, b0, (short)0, acc[0], false, false);
    acc[1] = __builtin_amdgcn_wmma_f32_16x16x32_f16(false, av, false, b1, (short)0, acc[1], false, false);
    acc[2] = __builtin_amdgcn_wmma_f32_16x16x32_f16(false, av, false, b2, (short)0, acc[2], false, false);
    acc[3] = __builtin_amdgcn_wmma_f32_16x16x32_f16(false, av, false, b3, (short)0, acc[3], false, false);
  }
  // D layout: VGPR r -> M = 16*tm + r + 8*half, N = 64*sn + 16*j + l15
  #pragma unroll
  for (int j = 0; j < 4; ++j) {
    int n = sn * 64 + j * 16 + l15;
    #pragma unroll
    for (int r = 0; r < 8; ++r) {
      int m = tm * 16 + r + half * 8;
      float v = acc[j][r];
      if (mode == 0) {
        v += p0[n];
      } else if (mode == 1) {
        v = (v + p0[n] - p1[n]) * rsqrtf(p2[n] + 1e-5f) * p3[n] + p4[n];
        v = v > 0.f ? v : 0.f;
      }
      C[(size_t)m * N + n] = v;
    }
  }
}

// ---------------------------------------------------------------------------
// GRU scan: one workgroup. W_hh (384x128 fp32, transposed, 192KB) lives in the
// 320KB CDNA5 LDS; the per-step gx row (1536B) is double-buffered in LDS via
// global_load_async_to_lds_b128 issued by spare threads one step ahead.
// ---------------------------------------------------------------------------
__global__ void gru_scan_kernel(const float* __restrict__ gx,     // [E,384]
                                const float* __restrict__ whh,    // [384,128]
                                const float* __restrict__ bhh,    // [384]
                                float* __restrict__ rnn)          // [E,128]
{
  extern __shared__ float smem[];
  float* sWt = smem;                  // [128*384], sWt[k*384+t] = whh[t*128+k]
  float* sGh = sWt + 128 * 384;       // [384]
  float* sH  = sGh + 384;             // [128]
  float* sGx = sH + 128;              // [2*384] double buffer for gx rows
  const int t = threadIdx.x;          // blockDim.x == 512

  for (int idx = t; idx < 384 * 128; idx += blockDim.x) {
    int r = idx / 128, k = idx % 128;
    sWt[k * 384 + r] = whh[idx];
  }
  if (t < 128) sH[t] = 0.f;
  float bh = (t < 384) ? bhh[t] : 0.f;
#if ASYNC_LDS
  if (t >= 384 && t < 480) {          // preload row 0 into buffer 0
    int p = t - 384;                  // 96 lanes x 16B = 1536B
    async_copy16(gx + p * 4, sGx + p * 4);
  }
  wait_async_zero();
#endif
  __syncthreads();

  for (int e = 0; e < EE; ++e) {
    if (t < 384) {
      float acc = bh;
      #pragma unroll 8
      for (int k = 0; k < 128; ++k)
        acc += sWt[k * 384 + t] * sH[k];   // sH[k] is an LDS broadcast
      sGh[t] = acc;
    } else {
      int p = t - 384;
#if ASYNC_LDS
      if (p < 96 && (e + 1) < EE)
        async_copy16(gx + (size_t)(e + 1) * 384 + p * 4,
                     sGx + ((e + 1) & 1) * 384 + p * 4);
      wait_async_zero();
#else
      if (p < 12 && (e + 1) < EE)
        __builtin_prefetch(gx + (size_t)(e + 1) * 384 + p * 32, 0, 1);
#endif
    }
    __syncthreads();
    if (t < 128) {
#if ASYNC_LDS
      const float* g = sGx + (e & 1) * 384;
#else
      const float* g = gx + (size_t)e * 384;
#endif
      float r  = 1.f / (1.f + __expf(-(g[t]       + sGh[t])));
      float z  = 1.f / (1.f + __expf(-(g[128 + t] + sGh[128 + t])));
      float c  = tanhf(g[256 + t] + r * sGh[256 + t]);
      float hn = (1.f - z) * c + z * sH[t];
      rnn[(size_t)e * 128 + t] = hn;
      sH[t] = hn;
    }
    __syncthreads();
  }
}

// ------------------------------ small kernels ------------------------------
__global__ void k_deg(const int* __restrict__ ea, float* __restrict__ deg) {
  int e = blockIdx.x * blockDim.x + threadIdx.x;
  if (e < EE) atomicAdd(&deg[ea[e]], 1.f);
}
__global__ void k_dinv(const float* __restrict__ deg, float* __restrict__ dinv) {
  int n = blockIdx.x * blockDim.x + threadIdx.x;
  if (n < NN) dinv[n] = deg[n] > 0.f ? rsqrtf(deg[n]) : 0.f;
}
__global__ void k_tx1(const int* __restrict__ ea, const float* __restrict__ dinv,
                      const float* __restrict__ x, float* __restrict__ tx1) {
  int idx = blockIdx.x * blockDim.x + threadIdx.x;
  if (idx >= EE * IN_) return;
  int e = idx >> 6, c = idx & 63;
  int s = ea[e], d = ea[EE + e];
  float w = -(dinv[s] * dinv[d]);
  atomicAdd(&tx1[(size_t)d * IN_ + c], w * x[(size_t)s * IN_ + c]);
}
__global__ void k_pack_wcat(const float* __restrict__ w0, const float* __restrict__ w1,
                            _Float16* __restrict__ dst) {
  int idx = blockIdx.x * blockDim.x + threadIdx.x;       // [128 n][128 k]
  if (idx >= 128 * 128) return;
  int n = idx >> 7, k = idx & 127;
  float v = (k < 64) ? w0[n * 64 + k] : w1[n * 64 + (k - 64)];
  dst[idx] = (_Float16)v;
}
__global__ void k_pack_f16(const float* __restrict__ src, _Float16* __restrict__ dst, int count) {
  int idx = blockIdx.x * blockDim.x + threadIdx.x;
  if (idx < count) dst[idx] = (_Float16)src[idx];
}
__global__ void k_pack_axcat(const float* __restrict__ x, const float* __restrict__ tx1,
                             _Float16* __restrict__ a) {
  int idx = blockIdx.x * blockDim.x + threadIdx.x;       // [N m][128 k]
  if (idx >= NN * 128) return;
  int m = idx >> 7, k = idx & 127;
  float v = (k < 64) ? x[(size_t)m * 64 + k] : tx1[(size_t)m * 64 + (k - 64)];
  a[idx] = (_Float16)v;
}
__global__ void k_build_ef(const int* __restrict__ ea, const float* __restrict__ h,
                           _Float16* __restrict__ ef) {
  int idx = blockIdx.x * blockDim.x + threadIdx.x;       // [E e][256 c]
  if (idx >= EE * 256) return;
  int e = idx >> 8, c = idx & 255;
  int node = (c < 128) ? ea[2 * e] : ea[2 * e + 1];      // interleaved view
  ef[idx] = (_Float16)h[(size_t)node * 128 + (c & 127)];
}
__global__ void k_cnt(const int* __restrict__ ea, float* __restrict__ cnt) {
  int idx = blockIdx.x * blockDim.x + threadIdx.x;
  if (idx < 2 * EE) atomicAdd(&cnt[ea[idx]], 1.f);
}
__global__ void k_agg(const int* __restrict__ ea, const float* __restrict__ rnn,
                      float* __restrict__ agg) {
  int idx = blockIdx.x * blockDim.x + threadIdx.x;
  if (idx >= EE * 128) return;
  int e = idx >> 7, j = idx & 127;
  float v = rnn[idx];
  atomicAdd(&agg[(size_t)ea[e] * 128 + j], v);
  atomicAdd(&agg[(size_t)ea[EE + e] * 128 + j], v);
}
__global__ void k_build_xc(const float* __restrict__ h, const float* __restrict__ agg,
                           const float* __restrict__ cnt, _Float16* __restrict__ xc) {
  int idx = blockIdx.x * blockDim.x + threadIdx.x;       // [N n][256 c]
  if (idx >= NN * 256) return;
  int n = idx >> 8, c = idx & 255;
  float v;
  if (c < 128) v = h[(size_t)n * 128 + c];
  else {
    float cn = cnt[n];
    if (cn == 0.f) cn = 1.f;
    v = agg[(size_t)n * 128 + (c - 128)] / cn;
  }
  xc[idx] = (_Float16)v;
}
__global__ void k_attdot(const float* __restrict__ xl, const float* __restrict__ as,
                         const float* __restrict__ ad, float* __restrict__ a_src,
                         float* __restrict__ a_dst) {
  int n = blockIdx.x * blockDim.x + threadIdx.x;
  if (n >= NN) return;
  float s = 0.f, d = 0.f;
  const float* row = xl + (size_t)n * 128;
  #pragma unroll 4
  for (int j = 0; j < 128; ++j) { s += row[j] * as[j]; d += row[j] * ad[j]; }
  a_src[n] = s; a_dst[n] = d;
}
__global__ void k_emax_init(const float* __restrict__ a_src, const float* __restrict__ a_dst,
                            float* __restrict__ emax) {
  int n = blockIdx.x * blockDim.x + threadIdx.x;
  if (n < NN) emax[n] = leaky02(a_src[n] + a_dst[n]);    // self-loop
}
__global__ void k_edge_max(const int* __restrict__ ea, const float* __restrict__ a_src,
                           const float* __restrict__ a_dst, float* __restrict__ emax) {
  int e = blockIdx.x * blockDim.x + threadIdx.x;
  if (e >= EE) return;
  int s = ea[e], d = ea[EE + e];
  atomicMaxF(&emax[d], leaky02(a_src[s] + a_dst[d]));
}
__global__ void k_den_init(const float* __restrict__ a_src, const float* __restrict__ a_dst,
                           const float* __restrict__ emax, float* __restrict__ den) {
  int n = blockIdx.x * blockDim.x + threadIdx.x;
  if (n < NN) den[n] = __expf(leaky02(a_src[n] + a_dst[n]) - emax[n]);
}
__global__ void k_edge_den(const int* __restrict__ ea, const float* __restrict__ a_src,
                           const float* __restrict__ a_dst, const float* __restrict__ emax,
                           float* __restrict__ den) {
  int e = blockIdx.x * blockDim.x + threadIdx.x;
  if (e >= EE) return;
  int s = ea[e], d = ea[EE + e];
  atomicAdd(&den[d], __expf(leaky02(a_src[s] + a_dst[d]) - emax[d]));
}
__global__ void k_alpha(const int* __restrict__ ea, const float* __restrict__ a_src,
                        const float* __restrict__ a_dst, const float* __restrict__ emax,
                        const float* __restrict__ den, float* __restrict__ alpha) {
  int e = blockIdx.x * blockDim.x + threadIdx.x;
  if (e >= EE) return;
  int s = ea[e], d = ea[EE + e];
  alpha[e] = __expf(leaky02(a_src[s] + a_dst[d]) - emax[d]) / den[d];
}
__global__ void k_gout_init(const float* __restrict__ a_src, const float* __restrict__ a_dst,
                            const float* __restrict__ emax, const float* __restrict__ den,
                            const float* __restrict__ xl, float* __restrict__ gout) {
  int idx = blockIdx.x * blockDim.x + threadIdx.x;
  if (idx >= NN * 128) return;
  int n = idx >> 7;
  float aself = __expf(leaky02(a_src[n] + a_dst[n]) - emax[n]) / den[n];
  gout[idx] = aself * xl[idx];
}
__global__ void k_edge_scatter(const int* __restrict__ ea, const float* __restrict__ alpha,
                               const float* __restrict__ xl, float* __restrict__ gout) {
  int idx = blockIdx.x * blockDim.x + threadIdx.x;
  if (idx >= EE * 128) return;
  int e = idx >> 7, j = idx & 127;
  int s = ea[e], d = ea[EE + e];
  atomicAdd(&gout[(size_t)d * 128 + j], alpha[e] * xl[(size_t)s * 128 + j]);
}
__global__ void k_pool(const float* __restrict__ gout, const float* __restrict__ gat_b,
                       const int* __restrict__ batch, float* __restrict__ pool) {
  int idx = blockIdx.x * blockDim.x + threadIdx.x;
  if (idx >= NN * 128) return;
  int n = idx >> 7, j = idx & 127;
  float v = gout[idx] + gat_b[j];
  v = v > 0.f ? v : 0.f;
  atomicAdd(&pool[(size_t)batch[n] * 128 + j], v);
}
__global__ void k_pcnt(const int* __restrict__ batch, float* __restrict__ pcnt) {
  int n = blockIdx.x * blockDim.x + threadIdx.x;
  if (n < NN) atomicAdd(&pcnt[batch[n]], 1.f);
}
__global__ void k_final(const float* __restrict__ pool, const float* __restrict__ pcnt,
                        const float* __restrict__ lin_w, const float* __restrict__ lin_b,
                        float* __restrict__ out) {
  int g = blockIdx.x * blockDim.x + threadIdx.x;
  if (g >= GG) return;
  float c = pcnt[g];
  if (c < 1.f) c = 1.f;
  float s = 0.f;
  for (int j = 0; j < 128; ++j) s += (pool[g * 128 + j] / c) * lin_w[j];
  out[g] = s + lin_b[0];
}

// ---------------------------------------------------------------------------
extern "C" void kernel_launch(void* const* d_in, const int* in_sizes, int n_in,
                              void* d_out, int out_size, void* d_ws, size_t ws_size,
                              hipStream_t stream)
{
  (void)in_sizes; (void)n_in; (void)out_size; (void)ws_size;
  const float* x        = (const float*)d_in[0];
  const int*   ea       = (const int*)d_in[1];
  const int*   batch    = (const int*)d_in[2];
  const float* cheb_w0  = (const float*)d_in[3];
  const float* cheb_w1  = (const float*)d_in[4];
  const float* cheb_b   = (const float*)d_in[5];
  const float* bn_gamma = (const float*)d_in[6];
  const float* bn_beta  = (const float*)d_in[7];
  const float* bn_mean  = (const float*)d_in[8];
  const float* bn_var   = (const float*)d_in[9];
  const float* gru_w_ih = (const float*)d_in[10];
  const float* gru_w_hh = (const float*)d_in[11];
  const float* gru_b_ih = (const float*)d_in[12];
  const float* gru_b_hh = (const float*)d_in[13];
  const float* gat_w    = (const float*)d_in[14];
  const float* att_src  = (const float*)d_in[15];
  const float* att_dst  = (const float*)d_in[16];
  const float* gat_b    = (const float*)d_in[17];
  const float* lin_w    = (const float*)d_in[18];
  const float* lin_b    = (const float*)d_in[19];
  float* out = (float*)d_out;

  char* ws = (char*)d_ws; size_t off = 0;
  auto carve = [&](size_t b) -> char* {
    char* p = ws + off; off = (off + b + 255) & ~(size_t)255; return p;
  };
  float* deg   = (float*)carve(NN * 4);
  float* dinv  = (float*)carve(NN * 4);
  float* cnt   = (float*)carve(NN * 4);
  float* asrc  = (float*)carve(NN * 4);
  float* adst  = (float*)carve(NN * 4);
  float* emax  = (float*)carve(NN * 4);
  float* den   = (float*)carve(NN * 4);
  float* alpha = (float*)carve(EE * 4);
  float* pool  = (float*)carve(GG * 128 * 4);
  float* pcnt  = (float*)carve(GG * 4);
  _Float16* wcat_t = (_Float16*)carve(128 * 128 * 2);
  _Float16* wih_t  = (_Float16*)carve(384 * 256 * 2);
  _Float16* wgat_t = (_Float16*)carve(128 * 256 * 2);
  float* h    = (float*)carve((size_t)NN * 128 * 4);
  float* xl   = (float*)carve((size_t)NN * 128 * 4);
  float* gout = (float*)carve((size_t)NN * 128 * 4);
  float* agg  = (float*)carve((size_t)NN * 128 * 4);
  float* gx   = (float*)carve((size_t)EE * 384 * 4);
  char*  big16 = carve((size_t)EE * 256 * 2);       // ef_f16, reused as rnn_out
  _Float16* ef_f16 = (_Float16*)big16;
  float*    rnn    = (float*)big16;                 // disjoint lifetime vs ef
  _Float16* axcat  = (_Float16*)carve((size_t)NN * 128 * 2);
  _Float16* xc_f16 = (_Float16*)carve((size_t)NN * 256 * 2);
  float* tx1 = (float*)carve((size_t)NN * IN_ * 4);

  // zero accumulation buffers (graph-capturable memset nodes)
  (void)hipMemsetAsync(deg,  0, NN * 4, stream);
  (void)hipMemsetAsync(cnt,  0, NN * 4, stream);
  (void)hipMemsetAsync(pool, 0, GG * 128 * 4, stream);
  (void)hipMemsetAsync(pcnt, 0, GG * 4, stream);
  (void)hipMemsetAsync(agg,  0, (size_t)NN * 128 * 4, stream);
  (void)hipMemsetAsync(tx1,  0, (size_t)NN * IN_ * 4, stream);

  auto blks = [](long n) { return (int)((n + 255) / 256); };

  // --- ChebConv ---
  k_deg <<<blks(EE), 256, 0, stream>>>(ea, deg);
  k_dinv<<<blks(NN), 256, 0, stream>>>(deg, dinv);
  k_tx1 <<<blks((long)EE * IN_), 256, 0, stream>>>(ea, dinv, x, tx1);
  k_pack_wcat <<<blks(128 * 128), 256, 0, stream>>>(cheb_w0, cheb_w1, wcat_t);
  k_pack_f16  <<<blks(384 * 256), 256, 0, stream>>>(gru_w_ih, wih_t, 384 * 256);
  k_pack_f16  <<<blks(128 * 256), 256, 0, stream>>>(gat_w, wgat_t, 128 * 256);
  k_pack_axcat<<<blks((long)NN * 128), 256, 0, stream>>>(x, tx1, axcat);
  // h = [x|tx1] @ Wcat^T + BN + ReLU  (M=8192,N=128,K=128 -> 1024 strips)
  wmma_gemm_kernel<<<1024 / 8, 256, 0, stream>>>(axcat, wcat_t, h,
      NN, 128, 128, 1, cheb_b, bn_mean, bn_var, bn_gamma, bn_beta);

  // --- GRU input gates ---
  k_build_ef<<<blks((long)EE * 256), 256, 0, stream>>>(ea, h, ef_f16);
  // gx = ef @ W_ih^T + b_ih (M=32768,N=384,K=256 -> 12288 strips)
  wmma_gemm_kernel<<<12288 / 8, 256, 0, stream>>>(ef_f16, wih_t, gx,
      EE, 384, 256, 0, gru_b_ih, nullptr, nullptr, nullptr, nullptr);

  // --- sequential GRU scan (W_hh + double-buffered gx rows in 320KB LDS) ---
  size_t smem = (size_t)(128 * 384 + 384 + 128 + 2 * 384) * 4;   // ~201.7 KB
  gru_scan_kernel<<<1, 512, smem, stream>>>(gx, gru_w_hh, gru_b_hh, rnn);

  // --- edge aggregation to nodes ---
  k_cnt<<<blks(2 * EE), 256, 0, stream>>>(ea, cnt);
  k_agg<<<blks((long)EE * 128), 256, 0, stream>>>(ea, rnn, agg);
  k_build_xc<<<blks((long)NN * 256), 256, 0, stream>>>(h, agg, cnt, xc_f16);

  // --- GAT linear: xl = xc @ gat_w^T (M=8192,N=128,K=256 -> 1024 strips) ---
  wmma_gemm_kernel<<<1024 / 8, 256, 0, stream>>>(xc_f16, wgat_t, xl,
      NN, 128, 256, 2, nullptr, nullptr, nullptr, nullptr, nullptr);

  // --- GAT attention softmax + aggregate (self-loops handled by init) ---
  k_attdot   <<<blks(NN), 256, 0, stream>>>(xl, att_src, att_dst, asrc, adst);
  k_emax_init<<<blks(NN), 256, 0, stream>>>(asrc, adst, emax);
  k_edge_max <<<blks(EE), 256, 0, stream>>>(ea, asrc, adst, emax);
  k_den_init <<<blks(NN), 256, 0, stream>>>(asrc, adst, emax, den);
  k_edge_den <<<blks(EE), 256, 0, stream>>>(ea, asrc, adst, emax, den);
  k_alpha    <<<blks(EE), 256, 0, stream>>>(ea, asrc, adst, emax, den, alpha);
  k_gout_init<<<blks((long)NN * 128), 256, 0, stream>>>(asrc, adst, emax, den, xl, gout);
  k_edge_scatter<<<blks((long)EE * 128), 256, 0, stream>>>(ea, alpha, xl, gout);

  // --- bias + relu + global mean pool + final linear ---
  k_pool <<<blks((long)NN * 128), 256, 0, stream>>>(gout, gat_b, batch, pool);
  k_pcnt <<<blks(NN), 256, 0, stream>>>(batch, pcnt);
  k_final<<<1, 32, 0, stream>>>(pool, pcnt, lin_w, lin_b, out);
}
